// GRU_75969381531776
// MI455X (gfx1250) — matrix operand order repaired and checked by reading, not compiled, not run on hardware
//
#include <hip/hip_runtime.h>
#include <cstddef>

typedef __attribute__((ext_vector_type(16))) _Float16 v16h;
typedef __attribute__((ext_vector_type(8)))  _Float16 v8h;
typedef __attribute__((ext_vector_type(8)))  float    v8f;

#define T_STEPS 512
#define BSZ 128
#define ISZ 256
#define HSZ 512
#define KSZ (HSZ + ISZ)   // 768
#define NBLK 64           // persistent grid size (all co-resident)

// ---------------------------------------------------------------------------
// Fragment builders (CDNA5 wave32 WMMA layouts, 16x16x32 f16)
// A (16x32, MxK): lane L holds row M=L%16; K = {0..7,16..23} (L<16) or
//                 {8..15,24..31} (L>=16), i.e. 16B at koff, 16B at koff+16.
// B (32x16, KxN): lane L holds col N=L%16; same K order as A, i.e. 16B at
//                 koff, 16B at koff+16.
// C (16x16 f32):  VGPR r, lane L: row = r + 8*(L>=16), col = L%16.
// ---------------------------------------------------------------------------

__device__ __forceinline__ v16h make_a_h16(const _Float16* __restrict__ p) {
  v8h a0 = *(const v8h*)p;
  v8h a1 = *(const v8h*)(p + 16);
  v16h a;
#pragma unroll
  for (int i = 0; i < 8; ++i) { a[i] = a0[i]; a[i + 8] = a1[i]; }
  return a;
}

__device__ __forceinline__ v16h make_a_f32(const float* __restrict__ p) {
  float t[16];
  *(float4*)(t + 0)  = *(const float4*)(p + 0);
  *(float4*)(t + 4)  = *(const float4*)(p + 4);
  *(float4*)(t + 8)  = *(const float4*)(p + 16);
  *(float4*)(t + 12) = *(const float4*)(p + 20);
  v16h a;
#pragma unroll
  for (int i = 0; i < 16; ++i) a[i] = (_Float16)t[i];
  return a;
}

__device__ __forceinline__ v16h make_b(const _Float16* __restrict__ wp) {
  v8h b0 = *(const v8h*)wp;         // 16B at koff, 16B at koff+16 (A-matching K order)
  v8h b1 = *(const v8h*)(wp + 16);
  return __builtin_shufflevector(b0, b1, 0, 1, 2, 3, 4, 5, 6, 7,
                                 8, 9, 10, 11, 12, 13, 14, 15);
}

// ---------------------------------------------------------------------------
// Device-wide generation barrier (all NBLK blocks resident by construction).
// Release fence before arrive (flush stores: global_wb), acquire fence after
// (invalidate WGP caches: global_inv) so zbuf/rh16/h16 handoffs are coherent.
// ---------------------------------------------------------------------------
__device__ __forceinline__ void grid_sync(unsigned* __restrict__ cnt,
                                          unsigned* __restrict__ gen) {
  __builtin_amdgcn_fence(__ATOMIC_RELEASE, "agent");
  __syncthreads();
  if (threadIdx.x == 0) {
    unsigned g = __hip_atomic_load(gen, __ATOMIC_RELAXED,
                                   __HIP_MEMORY_SCOPE_AGENT);
    unsigned a = __hip_atomic_fetch_add(cnt, 1u, __ATOMIC_ACQ_REL,
                                        __HIP_MEMORY_SCOPE_AGENT);
    if (a + 1u == (unsigned)NBLK) {
      __hip_atomic_store(cnt, 0u, __ATOMIC_RELAXED, __HIP_MEMORY_SCOPE_AGENT);
      __hip_atomic_store(gen, g + 1u, __ATOMIC_RELEASE,
                         __HIP_MEMORY_SCOPE_AGENT);
    } else {
      while (__hip_atomic_load(gen, __ATOMIC_ACQUIRE,
                               __HIP_MEMORY_SCOPE_AGENT) == g) {
        __builtin_amdgcn_s_sleep(2);
      }
    }
  }
  __syncthreads();
  __builtin_amdgcn_fence(__ATOMIC_ACQUIRE, "agent");
}

// ---------------------------------------------------------------------------
// Prologue kernels
// ---------------------------------------------------------------------------
__global__ __launch_bounds__(256) void cvt_weights(
    const float* __restrict__ Wz, const float* __restrict__ Wr,
    const float* __restrict__ Wh, _Float16* __restrict__ Wf16) {
  const int N = HSZ * KSZ;
  int i = blockIdx.x * blockDim.x + threadIdx.x;
  if (i < N)            Wf16[i] = (_Float16)Wz[i];
  else if (i < 2 * N)   Wf16[i] = (_Float16)Wr[i - N];
  else if (i < 3 * N)   Wf16[i] = (_Float16)Wh[i - 2 * N];
}

__global__ __launch_bounds__(256) void cvt_x16(
    const float* __restrict__ x, _Float16* __restrict__ x16, int n) {
  int i = blockIdx.x * blockDim.x + threadIdx.x;
  if (i < n) x16[i] = (_Float16)x[i];
}

__global__ __launch_bounds__(256) void zero_state(
    float* __restrict__ h, _Float16* __restrict__ h16, int n,
    unsigned* __restrict__ sync) {
  int i = blockIdx.x * blockDim.x + threadIdx.x;
  if (i < n) { h[i] = 0.0f; h16[i] = (_Float16)0.0f; }
  if (i < 2) sync[i] = 0u;
}

// ---------------------------------------------------------------------------
// Persistent GRU: one launch, T-step loop on-device, 2 grid syncs per step.
//
// Phase 1 (all 64 blocks, 8 waves each = 512 tile-waves):
//   block = one 16-col strip of the [B=128, 1024] (z|r) gate matrix; wave w
//   owns batch tile w.  g==0 -> zbuf = sigmoid(.), g==1 -> rh16 = f16(r*h).
// Phase 2 (all 64 blocks; 2 blocks per 16-col strip of [128, 512] cand):
//   wave pair shares one 16x16 tile, splitting K=768 in half; odd wave's
//   partial acc is reduced through LDS; even wave runs the epilogue:
//   cand = tanh(.), h_new = (1-z)*h + z*cand -> h ping-pong, h16, out[t].
// ---------------------------------------------------------------------------
template <bool USE_X16>
__global__ __launch_bounds__(256) void gru_persistent(
    const float* __restrict__ x, const _Float16* __restrict__ x16,
    const _Float16* __restrict__ Wf16,
    const float* __restrict__ bz, const float* __restrict__ br,
    const float* __restrict__ bh,
    float* __restrict__ hbuf0, float* __restrict__ hbuf1,
    _Float16* __restrict__ h16b0, _Float16* __restrict__ h16b1,
    float* __restrict__ zbuf, _Float16* __restrict__ rh16,
    float* __restrict__ out, unsigned* __restrict__ sync) {
  __shared__ float red[4][32][8];  // phase-2 partial-acc exchange

  const int blk  = blockIdx.x;     // 0..63
  const int lane = threadIdx.x & 31;
  const int wave = threadIdx.x >> 5;
  const int n    = lane & 15;      // A row / B col / C col
  const int ksel = lane >> 4;
  const int roff = ksel * 8;
  unsigned* cnt = sync;
  unsigned* gen = sync + 1;

  // ---- phase-1 invariants (hoisted out of the time loop) ----
  const int g1     = blk >> 5;               // 0 = z, 1 = r
  const int jcol1  = (blk & 31) << 4;        // 0..496 within gate
  const int bbase1 = wave << 4;
  const int jg1    = jcol1 + n;
  const float bias1 = g1 ? br[jg1] : bz[jg1];
  const _Float16* wrow1 = Wf16 + (size_t)g1 * HSZ * KSZ
                        + (size_t)(jcol1 + n) * KSZ + ksel * 8;

  // ---- phase-2 invariants ----
  const int jcol2  = (blk >> 1) << 4;        // 0..496
  const int bbase2 = ((((blk & 1) << 2) + (wave >> 1))) << 4;  // 0..112
  const int khalf  = wave & 1;               // K split: 0..383 / 384..767
  const int j2     = jcol2 + n;
  const float bias2 = bh[j2];
  const _Float16* Wh16  = Wf16 + (size_t)2 * HSZ * KSZ;
  const _Float16* wrow2 = Wh16 + (size_t)(jcol2 + n) * KSZ + ksel * 8;
  const _Float16* arow_rh = rh16 + (size_t)(bbase2 + n) * HSZ + ksel * 8;

  for (int t = 0; t < T_STEPS; ++t) {
    const float*    h_in    = (t & 1) ? hbuf1 : hbuf0;
    float*          h_out   = (t & 1) ? hbuf0 : hbuf1;
    const _Float16* h16_in  = (t & 1) ? h16b1 : h16b0;
    _Float16*       h16_out = (t & 1) ? h16b0 : h16b1;
    const size_t    xoff    = (size_t)t * BSZ * ISZ;

    // ================= phase 1 : z and r gates =================
    {
      const _Float16* arow_h = h16_in + (size_t)(bbase1 + n) * HSZ + ksel * 8;
      const _Float16* ax16 = x16 + xoff + (size_t)(bbase1 + n) * ISZ + ksel * 8;
      const float*    axf  = x   + xoff + (size_t)(bbase1 + n) * ISZ + ksel * 8;

      v8f acc = {};
#pragma unroll 4
      for (int kb = 0; kb < 16; ++kb) {               // K 0..511 (h-part)
        const int k0 = kb << 5;
        __builtin_prefetch(wrow1 + k0 + 128, 0, 1);
        v16h a = make_a_h16(arow_h + k0);
        v16h b = make_b(wrow1 + k0);
        acc = __builtin_amdgcn_wmma_f32_16x16x32_f16(false, a, false, b,
                                                     (short)0, acc, false, false);
      }
#pragma unroll 4
      for (int kb = 0; kb < 8; ++kb) {                // K 512..767 (x-part)
        const int k0 = kb << 5;
        v16h a = USE_X16 ? make_a_h16(ax16 + k0) : make_a_f32(axf + k0);
        v16h b = make_b(wrow1 + HSZ + k0);
        acc = __builtin_amdgcn_wmma_f32_16x16x32_f16(false, a, false, b,
                                                     (short)0, acc, false, false);
      }
#pragma unroll
      for (int r = 0; r < 8; ++r) {
        const int brow = bbase1 + roff + r;
        const float s  = 1.0f / (1.0f + __expf(-(acc[r] + bias1)));
        if (g1 == 0) {
          zbuf[(size_t)brow * HSZ + jg1] = s;
        } else {
          const float hv = h_in[(size_t)brow * HSZ + jg1];
          rh16[(size_t)brow * HSZ + jg1] = (_Float16)(s * hv);
        }
      }
    }

    grid_sync(cnt, gen);

    // ================= phase 2 : candidate + state update =================
    {
      const _Float16* ax16 = x16 + xoff + (size_t)(bbase2 + n) * ISZ + ksel * 8;
      const float*    axf  = x   + xoff + (size_t)(bbase2 + n) * ISZ + ksel * 8;

      v8f acc = {};
      if (khalf == 0) {
#pragma unroll 4
        for (int kb = 0; kb < 12; ++kb) {             // K 0..383
          const int k0 = kb << 5;
          __builtin_prefetch(wrow2 + k0 + 128, 0, 1);
          v16h a = make_a_h16(arow_rh + k0);
          v16h b = make_b(wrow2 + k0);
          acc = __builtin_amdgcn_wmma_f32_16x16x32_f16(false, a, false, b,
                                                       (short)0, acc, false, false);
        }
      } else {
#pragma unroll 4
        for (int kb = 12; kb < 16; ++kb) {            // K 384..511
          const int k0 = kb << 5;
          v16h a = make_a_h16(arow_rh + k0);
          v16h b = make_b(wrow2 + k0);
          acc = __builtin_amdgcn_wmma_f32_16x16x32_f16(false, a, false, b,
                                                       (short)0, acc, false, false);
        }
#pragma unroll 4
        for (int kb = 0; kb < 8; ++kb) {              // K 512..767 (x-part)
          const int k0 = kb << 5;
          v16h a = USE_X16 ? make_a_h16(ax16 + k0) : make_a_f32(axf + k0);
          v16h b = make_b(wrow2 + HSZ + k0);
          acc = __builtin_amdgcn_wmma_f32_16x16x32_f16(false, a, false, b,
                                                       (short)0, acc, false, false);
        }
        // hand partial acc to the even wave of the pair
        *(v8f*)red[wave >> 1][lane] = acc;
      }
      __syncthreads();
      if (khalf == 0) {
        v8f other = *(const v8f*)red[wave >> 1][lane];
#pragma unroll
        for (int r = 0; r < 8; ++r) {
          const int brow = bbase2 + roff + r;
          const float cand = tanhf(acc[r] + other[r] + bias2);
          const float zv   = zbuf[(size_t)brow * HSZ + j2];
          const float hv   = h_in[(size_t)brow * HSZ + j2];
          const float hnew = (1.0f - zv) * hv + zv * cand;
          h_out[(size_t)brow * HSZ + j2]   = hnew;
          h16_out[(size_t)brow * HSZ + j2] = (_Float16)hnew;
          out[(size_t)t * BSZ * HSZ + (size_t)brow * HSZ + j2] = hnew;
          if (t == T_STEPS - 1)  // h_last tail: [1,B,H] after [T,B,H]
            out[(size_t)T_STEPS * BSZ * HSZ + (size_t)brow * HSZ + j2] = hnew;
        }
      }
    }

    grid_sync(cnt, gen);
  }
}

// ---------------------------------------------------------------------------
// Host: 3-4 launches total
// ---------------------------------------------------------------------------
extern "C" void kernel_launch(void* const* d_in, const int* in_sizes, int n_in,
                              void* d_out, int out_size, void* d_ws, size_t ws_size,
                              hipStream_t stream) {
  (void)in_sizes; (void)n_in; (void)out_size;
  const float* x  = (const float*)d_in[0];
  const float* Wz = (const float*)d_in[1];
  const float* bz = (const float*)d_in[2];
  const float* Wr = (const float*)d_in[3];
  const float* br = (const float*)d_in[4];
  const float* Wh = (const float*)d_in[5];
  const float* bh = (const float*)d_in[6];
  float* out = (float*)d_out;

  // workspace layout (bytes):
  //   [0)          Wf16:  3*512*768*2 = 2,359,296
  //   [2,359,296)  hbuf0: 262,144        [2,621,440)  hbuf1: 262,144
  //   [2,883,584)  h16b0: 131,072        [3,014,656)  h16b1: 131,072
  //   [3,145,728)  zbuf:  262,144        [3,407,872)  rh16:  131,072
  //   [3,538,944)  sync:  64
  //   [3,539,008)  x16:   33,554,432 (optional)
  char* ws = (char*)d_ws;
  _Float16* Wf16  = (_Float16*)ws;
  float*    hbuf0 = (float*)(ws + 2359296);
  float*    hbuf1 = (float*)(ws + 2621440);
  _Float16* h16b0 = (_Float16*)(ws + 2883584);
  _Float16* h16b1 = (_Float16*)(ws + 3014656);
  float*    zbuf  = (float*)(ws + 3145728);
  _Float16* rh16  = (_Float16*)(ws + 3407872);
  unsigned* sync  = (unsigned*)(ws + 3538944);
  _Float16* x16   = (_Float16*)(ws + 3539008);

  const size_t NEED_X16 = 3539008 + (size_t)T_STEPS * BSZ * ISZ * 2;
  const bool useX16 = ws_size >= NEED_X16;

  const int NW = 3 * HSZ * KSZ;  // 1,179,648
  cvt_weights<<<(NW + 255) / 256, 256, 0, stream>>>(Wz, Wr, Wh, Wf16);
  zero_state<<<(BSZ * HSZ + 255) / 256, 256, 0, stream>>>(hbuf0, h16b0,
                                                          BSZ * HSZ, sync);
  if (useX16) {
    const int NX = T_STEPS * BSZ * ISZ;  // 16,777,216
    cvt_x16<<<(NX + 255) / 256, 256, 0, stream>>>(x, x16, NX);
    gru_persistent<true><<<NBLK, 256, 0, stream>>>(
        x, x16, Wf16, bz, br, bh, hbuf0, hbuf1, h16b0, h16b1,
        zbuf, rh16, out, sync);
  } else {
    gru_persistent<false><<<NBLK, 256, 0, stream>>>(
        x, x16, Wf16, bz, br, bh, hbuf0, hbuf1, h16b0, h16b1,
        zbuf, rh16, out, sync);
  }
}
